// RetinaNet_87462714016343
// MI455X (gfx1250) — matrix operand order repaired and checked by reading, not compiled, not run on hardware
//
#include <hip/hip_runtime.h>
#include <hip/hip_bf16.h>

typedef __attribute__((ext_vector_type(16))) __bf16 v16bf;
typedef __attribute__((ext_vector_type(8)))  float  v8f;
typedef __attribute__((vector_size(16)))     int    v4i;

#define GN_NP 64   // partial blocks per (b,g) for group-norm stats

#if defined(__AMDGCN__) && __has_builtin(__builtin_amdgcn_global_load_async_to_lds_b128) && \
    __has_builtin(__builtin_amdgcn_s_wait_asynccnt)
#define USE_ASYNC_STAGE 1
#else
#define USE_ASYNC_STAGE 0
#endif

static __device__ __forceinline__ __bf16 f2bf(float f) {
    union { float f; unsigned u; } un; un.f = f;
    unsigned r = un.u + 0x7FFFu + ((un.u >> 16) & 1u);   // round-to-nearest-even
    unsigned short h = (unsigned short)(r >> 16);
    return __builtin_bit_cast(__bf16, h);
}

// NCDHW f32 -> NDHWC bf16
__global__ void to_ndhwc_bf16(const float* __restrict__ in, __bf16* __restrict__ out,
                              int B, int sp) {
    int idx = blockIdx.x * blockDim.x + threadIdx.x;
    int total = B * sp * 128;
    if (idx >= total) return;
    int c = idx & 127, m = idx >> 7;
    int b = m / sp, s = m % sp;
    out[idx] = f2bf(in[((size_t)b * 128 + c) * sp + s]);
}

// f32 weights [O][I=128][27taps] -> bf16 [tap][Co16][128], zero-padded O
__global__ void repack_w(const float* __restrict__ w, __bf16* __restrict__ out,
                         int Co, int Co16) {
    int idx = blockIdx.x * blockDim.x + threadIdx.x;
    int total = 27 * Co16 * 128;
    if (idx >= total) return;
    int i = idx & 127, rest = idx >> 7;
    int o = rest % Co16, t = rest / Co16;
    float v = 0.0f;
    if (o < Co) v = w[((size_t)o * 128 + i) * 27 + t];
    out[idx] = f2bf(v);
}

// Implicit-GEMM 3x3x3 SAME conv, Cin=128, via v_wmma_f32_16x16x32_bf16.
// act: NDHWC bf16; wgt: [tap][Co16][128] bf16.
// 16*MT output rows per workgroup; one wave per 16 output channels.
// Double-buffered LDS A tile staged with async global->LDS loads (ASYNCcnt),
// waited only after the current tap's WMMA block. Spatial decode hoisted out
// of the tap loop.
// out: ncdhw ? NCDHW f32 (final, d_out) : [M][128] f32 (intermediate, ws)
template<int MT, int NQ>
__global__ void conv3d_wmma_bf16(const __bf16* __restrict__ act,
                                 const __bf16* __restrict__ wgt,
                                 const float* __restrict__ bias,
                                 float* __restrict__ out,
                                 int B, int D, int H, int W,
                                 int Co16, int Co, int ncdhw) {
    const int lane  = threadIdx.x & 31;
    const int wave  = threadIdx.x >> 5;
    const int HW    = H * W, sp = D * HW;
    const int Mtot  = B * sp;
    const int mbase = blockIdx.x * (16 * MT);
    const int nbase = wave << 4;
    const int ln15  = lane & 15;
    const int khalf = lane >> 4;

    __shared__ __align__(16) __bf16 As[2][MT * 16 * 128];   // double-buffered A tile

    // --- hoisted, tap-invariant staging decode: chunk q -> (valid, z,y,x, bases)
    bool sval[NQ];
    int  sz[NQ], sy[NQ], sx[NQ], sbase[NQ], sdst[NQ];
#pragma unroll
    for (int iq = 0; iq < NQ; ++iq) {
        sval[iq] = false; sz[iq] = sy[iq] = sx[iq] = sbase[iq] = sdst[iq] = 0;
        const int q = (int)threadIdx.x + iq * (int)blockDim.x;
        if (q < MT * 16 * 16) {
            const int row = q >> 4, cc = q & 15;
            sdst[iq] = row * 128 + cc * 8;
            const int m = mbase + row;
            if (m < Mtot) {
                const int b = m / sp, s = m % sp;
                sz[iq] = s / HW;
                const int r2 = s % HW;
                sy[iq] = r2 / W;
                sx[iq] = r2 % W;
                sbase[iq] = b * sp * 128 + cc * 8;
                sval[iq] = true;
            }
        }
    }

    // stage A tile for tap t into buffer bufi (zero-pad borders & rows >= Mtot)
    auto stage = [&](int t, int bufi) {
        const int kd = t / 9 - 1, kh = (t / 3) % 3 - 1, kw = t % 3 - 1;
#pragma unroll
        for (int iq = 0; iq < NQ; ++iq) {
            const int q = (int)threadIdx.x + iq * (int)blockDim.x;
            if (q >= MT * 16 * 16) break;
            bool   ok   = false;
            size_t goff = 0;
            if (sval[iq]) {
                const int iz = sz[iq] + kd, iy = sy[iq] + kh, ix = sx[iq] + kw;
                if (iz >= 0 && iz < D && iy >= 0 && iy < H && ix >= 0 && ix < W) {
                    ok   = true;
                    goff = (size_t)sbase[iq] + (size_t)((iz * H + iy) * W + ix) * 128;
                }
            }
#if USE_ASYNC_STAGE
            if (ok) {
                // direct global -> LDS copy, tracked by ASYNCcnt (no VGPR transit)
                __builtin_amdgcn_global_load_async_to_lds_b128(
                    (v4i*)(act + goff), (v4i*)(As[bufi] + sdst[iq]), 0, 0);
            } else {
                *(uint4*)(As[bufi] + sdst[iq]) = uint4{0u, 0u, 0u, 0u};
            }
#else
            uint4 v = {0u, 0u, 0u, 0u};
            if (ok) v = *(const uint4*)(act + goff);
            *(uint4*)(As[bufi] + sdst[iq]) = v;
#endif
        }
    };

    v8f acc[MT];
#pragma unroll
    for (int st = 0; st < MT; ++st) acc[st] = v8f{};

    stage(0, 0);
#if USE_ASYNC_STAGE
    __builtin_amdgcn_s_wait_asynccnt(0);
#endif
    __syncthreads();

    for (int t = 0; t < 27; ++t) {
        if (t + 1 < 27) {
            // prefetch next tap's weight block for this wave (global_prefetch)
            __builtin_prefetch(wgt + ((size_t)(t + 1) * Co16 + nbase) * 128, 0, 3);
            stage(t + 1, (t + 1) & 1);
        }
        const __bf16* wt   = wgt + ((size_t)t * Co16 + nbase + ln15) * 128;
        const __bf16* abuf = As[t & 1];
#pragma unroll
        for (int ck = 0; ck < 4; ++ck) {
            const int kb = ck << 5;
            // B 32x16 bf16: lane holds N=ln15, 16 contiguous K at kb+khalf*16
            v16bf bm;
            const __bf16* bp = wt + kb + khalf * 16;
            ((uint4*)&bm)[0] = *(const uint4*)bp;
            ((uint4*)&bm)[1] = *(const uint4*)(bp + 8);
#pragma unroll
            for (int st = 0; st < MT; ++st) {
                // A 16x32 bf16: lane holds M=ln15, 16 contiguous K at kb+khalf*16
                v16bf a;
                const __bf16* ap = abuf + (st * 16 + ln15) * 128 + kb + khalf * 16;
                ((uint4*)&a)[0] = *(const uint4*)ap;
                ((uint4*)&a)[1] = *(const uint4*)(ap + 8);
                acc[st] = __builtin_amdgcn_wmma_f32_16x16x32_bf16(false, a, false, bm,
                                                                  (short)0, acc[st],
                                                                  false, false);
            }
        }
#if USE_ASYNC_STAGE
        __builtin_amdgcn_s_wait_asynccnt(0);   // next-tap staging landed in LDS
#endif
        __syncthreads();
    }

    const int co = nbase + ln15;
    if (co < Co) {
        const float bv = bias[co];
#pragma unroll
        for (int st = 0; st < MT; ++st) {
#pragma unroll
            for (int r = 0; r < 8; ++r) {
                const int m = mbase + st * 16 + r + (khalf << 3);
                if (m < Mtot) {
                    const float val = acc[st][r] + bv;
                    if (ncdhw) {
                        const int b = m / sp, s = m % sp;
                        out[((size_t)b * Co + co) * sp + s] = val;
                    } else {
                        out[(size_t)m * 128 + co] = val;
                    }
                }
            }
        }
    }
}

// GroupNorm stats stage 1: grid = B*8*GN_NP blocks; partial {sum, sumsq} per block.
__global__ void gn_stats_part(const float* __restrict__ conv, float* __restrict__ part,
                              int sp) {
    const int blk = blockIdx.x, p = blk % GN_NP, bg = blk / GN_NP;
    const int b = bg >> 3, g = bg & 7;
    const int n = sp * 16;
    float s0 = 0.0f, s1 = 0.0f;
    for (int i = p * (int)blockDim.x + (int)threadIdx.x; i < n; i += GN_NP * (int)blockDim.x) {
        const int s = i >> 4, c = (i & 15) + (g << 4);
        const float v = conv[(size_t)(b * sp + s) * 128 + c];
        s0 += v; s1 += v * v;
    }
    __shared__ float r0[256], r1[256];
    r0[threadIdx.x] = s0; r1[threadIdx.x] = s1;
    __syncthreads();
    for (int st = 128; st > 0; st >>= 1) {
        if (threadIdx.x < st) { r0[threadIdx.x] += r0[threadIdx.x + st];
                                r1[threadIdx.x] += r1[threadIdx.x + st]; }
        __syncthreads();
    }
    if (threadIdx.x == 0) {
        part[(bg * GN_NP + p) * 2]     = r0[0];
        part[(bg * GN_NP + p) * 2 + 1] = r1[0];
    }
}

// GroupNorm stats stage 2: grid = B*8 blocks of GN_NP threads; stats[bg] = {mean, rstd}
__global__ void gn_stats_final(const float* __restrict__ part, float* __restrict__ stats,
                               int sp) {
    const int bg = blockIdx.x;
    __shared__ float r0[GN_NP], r1[GN_NP];
    r0[threadIdx.x] = part[(bg * GN_NP + threadIdx.x) * 2];
    r1[threadIdx.x] = part[(bg * GN_NP + threadIdx.x) * 2 + 1];
    __syncthreads();
    for (int st = GN_NP / 2; st > 0; st >>= 1) {
        if ((int)threadIdx.x < st) { r0[threadIdx.x] += r0[threadIdx.x + st];
                                     r1[threadIdx.x] += r1[threadIdx.x + st]; }
        __syncthreads();
    }
    if (threadIdx.x == 0) {
        const float n    = (float)(sp * 16);
        const float mean = r0[0] / n;
        const float var  = r1[0] / n - mean * mean;
        stats[bg * 2]     = mean;
        stats[bg * 2 + 1] = rsqrtf(var + 1e-5f);
    }
}

// normalize + affine + ReLU + f32->bf16 (NDHWC layout preserved)
__global__ void gn_apply_relu_bf16(const float* __restrict__ conv, const float* __restrict__ stats,
                                   const float* __restrict__ gamma, const float* __restrict__ beta,
                                   __bf16* __restrict__ actout, int B, int sp) {
    int idx = blockIdx.x * blockDim.x + threadIdx.x;
    int total = B * sp * 128;
    if (idx >= total) return;
    const int c = idx & 127, m = idx >> 7;
    const int b = m / sp, g = c >> 4;
    const float mean = stats[(b * 8 + g) * 2];
    const float rstd = stats[(b * 8 + g) * 2 + 1];
    float v = (conv[idx] - mean) * rstd * gamma[c] + beta[c];
    actout[idx] = f2bf(fmaxf(v, 0.0f));
}

extern "C" void kernel_launch(void* const* d_in, const int* in_sizes, int n_in,
                              void* d_out, int out_size, void* d_ws, size_t ws_size,
                              hipStream_t stream) {
    (void)in_sizes; (void)n_in; (void)out_size; (void)ws_size;
    const float* feat[2] = { (const float*)d_in[0], (const float*)d_in[1] };

    // param indices: base 2 + head*18; conv i: w=+4i,b,g,be; out: wo=+16, bo=+17
    const float *wc[2][4], *bc[2][4], *gc[2][4], *bec[2][4], *wo[2], *bo[2];
    for (int h = 0; h < 2; ++h) {
        int base = 2 + h * 18;
        for (int i = 0; i < 4; ++i) {
            wc[h][i]  = (const float*)d_in[base + 4 * i + 0];
            bc[h][i]  = (const float*)d_in[base + 4 * i + 1];
            gc[h][i]  = (const float*)d_in[base + 4 * i + 2];
            bec[h][i] = (const float*)d_in[base + 4 * i + 3];
        }
        wo[h] = (const float*)d_in[base + 16];
        bo[h] = (const float*)d_in[base + 17];
    }
    const int CoH[2] = { 15, 18 }, Co16H[2] = { 16, 32 };

    // workspace layout
    size_t off = 0;
    auto alloc = [&](size_t bytes) -> void* {
        off = (off + 255) & ~(size_t)255;
        void* p = (char*)d_ws + off;
        off += bytes;
        return p;
    };
    const int Mmax = 2 * 20 * 20 * 20;                   // 16000
    __bf16* act0   = (__bf16*)alloc((size_t)Mmax * 128 * 2);
    __bf16* actP   = (__bf16*)alloc((size_t)Mmax * 128 * 2);
    __bf16* actQ   = (__bf16*)alloc((size_t)Mmax * 128 * 2);
    float*  convbuf= (float*) alloc((size_t)Mmax * 128 * 4);
    float*  stats  = (float*) alloc(16 * 2 * 4);
    float*  part   = (float*) alloc(16 * GN_NP * 2 * 4);
    __bf16* wr[2][5];
    for (int h = 0; h < 2; ++h) {
        for (int i = 0; i < 4; ++i)
            wr[h][i] = (__bf16*)alloc((size_t)27 * 128 * 128 * 2);
        wr[h][4] = (__bf16*)alloc((size_t)27 * Co16H[h] * 128 * 2);
    }

    // repack all weights once (bf16, [tap][Co16][Cin])
    for (int h = 0; h < 2; ++h) {
        for (int i = 0; i < 4; ++i) {
            int total = 27 * 128 * 128;
            repack_w<<<(total + 255) / 256, 256, 0, stream>>>(wc[h][i], wr[h][i], 128, 128);
        }
        int total = 27 * Co16H[h] * 128;
        repack_w<<<(total + 255) / 256, 256, 0, stream>>>(wo[h], wr[h][4], CoH[h], Co16H[h]);
    }

    float* outp = (float*)d_out;
    float* outs[2][2] = { { outp + 0,      outp + 240000 },    // cls0, cls1
                          { outp + 270000, outp + 558000 } };  // reg0, reg1

    const int Dims[2] = { 20, 10 };
    for (int f = 0; f < 2; ++f) {
        const int D = Dims[f], H = D, W = D, B = 2;
        const int sp = D * H * W, M = B * sp;
        const int grid64 = (M + 63) / 64;
        // f32 NCDHW -> bf16 NDHWC
        to_ndhwc_bf16<<<(M * 128 + 255) / 256, 256, 0, stream>>>(feat[f], act0, B, sp);
        for (int h = 0; h < 2; ++h) {
            const __bf16* cur = act0;
            __bf16* ping = actP; __bf16* pong = actQ;
            for (int l = 0; l < 4; ++l) {
                conv3d_wmma_bf16<4, 4><<<grid64, 256, 0, stream>>>(cur, wr[h][l], bc[h][l],
                                                                   convbuf, B, D, H, W,
                                                                   128, 128, 0);
                gn_stats_part<<<B * 8 * GN_NP, 256, 0, stream>>>(convbuf, part, sp);
                gn_stats_final<<<B * 8, GN_NP, 0, stream>>>(part, stats, sp);
                gn_apply_relu_bf16<<<(M * 128 + 255) / 256, 256, 0, stream>>>(
                    convbuf, stats, gc[h][l], bec[h][l], ping, B, sp);
                cur = ping;
                __bf16* tmp = ping; ping = pong; pong = tmp;
            }
            conv3d_wmma_bf16<1, 8><<<M / 16, 2 * Co16H[h], 0, stream>>>(cur, wr[h][4], bo[h],
                                                                        outs[h][f],
                                                                        B, D, H, W,
                                                                        Co16H[h], CoH[h], 1);
        }
    }
}
